// AngleFreqEnhance_41326175322550
// MI455X (gfx1250) — compile-verified, matrix operand order checked by reading, not run on hardware
//
#include <hip/hip_runtime.h>
#include <hip/hip_bf16.h>
#include <math.h>

// ---------------------------------------------------------------------------
// AngleFreqEnhance for MI455X (gfx1250, wave32)
//   out = x + W_out @ IFFT2( G .* FFT2( W_in @ x ) )
// B=4, C=256, H=W=256, c_mid=16, planes = B*c_mid = 64, HW = 65536
// Memory-bound (~768 MiB of HBM traffic -> ~33us floor at 23.3 TB/s);
// channel mixes use bf16 WMMA, x is staged to LDS via the Tensor Data Mover.
// ---------------------------------------------------------------------------

typedef __attribute__((ext_vector_type(16))) __bf16    v16bf;
typedef __attribute__((ext_vector_type(8)))  float     v8f;
typedef __attribute__((ext_vector_type(4)))  unsigned  v4u;
typedef __attribute__((ext_vector_type(8)))  unsigned  v8u;

#define PI_F 3.14159265358979f
#define TWO_PI_F 6.28318530717959f

// CDNA5 16-bit A/B matrix K index for (vgpr v, half h, lane>=16)
__device__ __forceinline__ int kmap(int v, int h, int laneHi) {
    return ((v & 4) << 2) + (laneHi << 3) + ((v & 3) << 1) + h;
}

// ---------------------------------------------------------------------------
// TDM: DMA a 32(ch) x 128(px) f32 tile from global into LDS.
// D# per CDNA5 ISA ch.8: group0 = {flags, lds_addr, global_addr, type},
// group1 = {data_size/pad, dims, tile dims, strides}.
// Pad: after every 128 dwords (interval code 6) insert 1 dword (amount 0)
// -> LDS row stride 129 dwords (bank-conflict-free half-wave reads).
// ---------------------------------------------------------------------------
__device__ __forceinline__ void tdm_load_tile(const float* gptr, unsigned lds_off) {
    const unsigned long long ga = (unsigned long long)(uintptr_t)gptr;
    v4u g0;
    g0[0] = 1u;                                             // count=1, user mode
    g0[1] = lds_off;                                        // LDS byte address
    g0[2] = (unsigned)ga;                                   // global addr [31:0]
    g0[3] = ((unsigned)(ga >> 32) & 0x01FFFFFFu)            // global addr [56:32]
          | (2u << 30);                                     // type = 2 (image)
    v8u g1;
    g1[0] = (2u << 16)                                      // data_size = 4 bytes
          | (1u << 20)                                      // pad_enable
          | (6u << 22);                                     // pad_interval: 128 dwords
                                                            // pad_amount = 0 -> 1 dword
    g1[1] = 0u;                                             // tensor_dim0(=65536) lo16 in [31:16] = 0
    g1[2] = 1u                                              // tensor_dim0 hi bits -> 65536
          | (256u << 16);                                   // tensor_dim1 = 256 (lo16)
    g1[3] = (128u << 16);                                   // tensor_dim1 hi=0 | tile_dim0 = 128
    g1[4] = 32u;                                            // tile_dim1 = 32, tile_dim2 = 0
    g1[5] = 65536u;                                         // tensor_dim0_stride lo32
    g1[6] = 0u;                                             // stride hi | dim1_stride lo
    g1[7] = 0u;                                             // dim1_stride hi (unused, 2D tile)
    asm volatile("tensor_load_to_lds %0, %1" :: "s"(g0), "s"(g1) : "memory");
}

// ---------------------------------------------------------------------------
// Kernel 1: xp[b,m,hw] = sum_c w_in[m,c] * x[b,c,hw]   (WMMA bf16, K=256)
// Block = 8 waves = 128 pixels; TDM double-buffers 32-channel tiles in LDS.
// ---------------------------------------------------------------------------
__global__ void __launch_bounds__(256)
proj_in_kernel(const float* __restrict__ x, const float* __restrict__ w_in,
               float* __restrict__ xp) {
    __shared__ float stage[2][32 * 129];                    // 2 x 16.1 KB
    const int tid    = threadIdx.x;
    const int lane   = tid & 31;
    const int wave   = tid >> 5;
    const int row    = lane & 15;                           // A row (m) / B col (px)
    const int laneHi = lane >> 4;
    const int p0     = blockIdx.x * 128;                    // block pixel base
    const int b      = p0 >> 16;                            // 65536 px per batch
    const int hw0    = p0 & 65535;
    const int pl     = wave * 16 + row;                     // pixel within tile
    const int hw     = hw0 + pl;

    if (wave == 0)
        tdm_load_tile(x + (((size_t)b * 256) << 16) + hw0,
                      (unsigned)(uintptr_t)&stage[0][0]);

    v8f acc = {};
    #pragma unroll
    for (int chunk = 0; chunk < 8; ++chunk) {
        if (wave == 0) {
            if (chunk < 7) {
                tdm_load_tile(x + (((size_t)b * 256 + (chunk + 1) * 32) << 16) + hw0,
                              (unsigned)(uintptr_t)&stage[(chunk + 1) & 1][0]);
                __builtin_amdgcn_s_wait_tensorcnt(1);       // chunk's tile landed
            } else {
                __builtin_amdgcn_s_wait_tensorcnt(0);
            }
        }
        __syncthreads();                                    // tile visible to all waves

        const float* sb = &stage[chunk & 1][0];
        v16bf A, Bm;
        #pragma unroll
        for (int v = 0; v < 8; ++v) {
            #pragma unroll
            for (int h = 0; h < 2; ++h) {
                const int k = kmap(v, h, laneHi);
                A[2 * v + h]  = (__bf16)w_in[row * 256 + chunk * 32 + k];
                Bm[2 * v + h] = (__bf16)sb[k * 129 + pl];
            }
        }
        acc = __builtin_amdgcn_wmma_f32_16x16x32_bf16(
            false, A, false, Bm, (short)0, acc, false, false);
        __syncthreads();                                    // before buffer reuse
    }
    const int mbase = laneHi << 3;
    #pragma unroll
    for (int d = 0; d < 8; ++d)
        xp[(((size_t)b * 16 + mbase + d) << 16) + hw] = acc[d];
}

// ---------------------------------------------------------------------------
// Shared radix-2 DIT FFT over 8 lines of 256 complex values in LDS.
// 256 threads cooperate; input must be stored bit-reversed.
// ---------------------------------------------------------------------------
__device__ __forceinline__ void fft8lines(float2 (*s)[256], int t, float sign) {
    for (int st = 1; st <= 8; ++st) {
        const int len  = 1 << st;
        const int half = len >> 1;
        __syncthreads();
        #pragma unroll
        for (int i = 0; i < 4; ++i) {
            const int idx  = t + i * 256;          // 0..1023 = 8 lines * 128 bflys
            const int line = idx >> 7;
            const int bf   = idx & 127;
            const int grp  = bf >> (st - 1);
            const int pos  = bf & (half - 1);
            const int i0   = grp * len + pos;
            const int i1   = i0 + half;
            const float ang = sign * TWO_PI_F * (float)pos / (float)len;
            float sn, cs;
            __sincosf(ang, &sn, &cs);
            const float2 a  = s[line][i0];
            const float2 bb = s[line][i1];
            const float2 tb = make_float2(fmaf(bb.x, cs, -bb.y * sn),
                                          fmaf(bb.x, sn,  bb.y * cs));
            s[line][i0] = make_float2(a.x + tb.x, a.y + tb.y);
            s[line][i1] = make_float2(a.x - tb.x, a.y - tb.y);
        }
    }
    __syncthreads();
}

// ---------------------------------------------------------------------------
// Kernel 2: forward FFT along rows (real input), ortho scale 1/16 per pass
// ---------------------------------------------------------------------------
__global__ void __launch_bounds__(256)
fft_rows_fwd_kernel(const float* __restrict__ xp, float2* __restrict__ spec) {
    __shared__ float2 tile[8][256];
    const int blk   = blockIdx.x;
    const int plane = blk >> 5;
    const int row0  = (blk & 31) * 8;
    const size_t base = ((size_t)plane << 16) + ((size_t)row0 << 8);
    const int t = threadIdx.x;

    #pragma unroll
    for (int i = 0; i < 8; ++i) {
        const int idx = t + i * 256;
        const int l = idx >> 8, k = idx & 255;
        tile[l][__brev((unsigned)k) >> 24] = make_float2(xp[base + idx], 0.0f);
    }
    fft8lines(tile, t, -1.0f);
    #pragma unroll
    for (int i = 0; i < 8; ++i) {
        const int idx = t + i * 256;
        const int l = idx >> 8, k = idx & 255;
        const float2 v = tile[l][k];
        spec[base + idx] = make_float2(v.x * 0.0625f, v.y * 0.0625f);
    }
}

// ---------------------------------------------------------------------------
// HF ring gain at shifted frequency coordinate (fftshift folded via k^128)
// ---------------------------------------------------------------------------
__device__ float compute_gain(const float* __restrict__ weights, int m,
                              int ky, int kx, int n_radii) {
    const int ys = ky ^ 128, xs = kx ^ 128;
    const float dy = (float)ys - 128.0f, dx = (float)xs - 128.0f;
    const float r = sqrtf(dy * dy + dx * dx);
    if (!(r > 0.8f * 128.0f)) return 1.0f;             // low-freq: gain 1
    const float theta = fmodf(atan2f(dy, dx) + PI_F, PI_F);
    int ridx = (int)floorf(r * 0.125f);                // r / RADIUS_WIDTH
    ridx = min(ridx, n_radii - 1);
    const float delta = PI_F / 8.0f;
    const float hwid  = 1.5f * delta * 0.5f;
    float aw[8], ssum = 0.0f;
    #pragma unroll
    for (int a = 0; a < 8; ++a) {
        const float center = (float)a * delta + 0.5f * delta;
        const float d = fabsf(theta - center);
        const float w = (d < hwid) ? fmaxf(1.0f - d / hwid, 0.0f) : 0.0f;
        aw[a] = w;
        ssum += w;
    }
    const float inv = 1.0f / (ssum + 1e-8f);
    float g = 0.0f;
    #pragma unroll
    for (int a = 0; a < 8; ++a)
        g += aw[a] * inv * weights[(m * 8 + a) * n_radii + ridx];
    return 1.0f + 0.5f * tanhf(g);
}

// ---------------------------------------------------------------------------
// Kernel 3: column FFT -> gain -> column IFFT, fused in LDS (8 cols/block)
// ---------------------------------------------------------------------------
__global__ void __launch_bounds__(256)
fft_cols_gain_kernel(float2* __restrict__ spec, const float* __restrict__ weights,
                     int n_radii) {
    __shared__ float2 t0[8][256];
    __shared__ float2 t1[8][256];
    const int blk   = blockIdx.x;
    const int plane = blk >> 5;
    const int kx0   = (blk & 31) * 8;
    const int m     = plane & 15;
    const size_t base = (size_t)plane << 16;
    const int t = threadIdx.x;

    #pragma unroll
    for (int i = 0; i < 8; ++i) {
        const int idx = t + i * 256;
        const int ky = idx >> 3, c = idx & 7;
        t0[c][__brev((unsigned)ky) >> 24] = spec[base + ((size_t)ky << 8) + kx0 + c];
    }
    fft8lines(t0, t, -1.0f);                 // forward along columns
    #pragma unroll
    for (int i = 0; i < 8; ++i) {
        const int idx = t + i * 256;
        const int ky = idx >> 3, c = idx & 7;
        const float g = compute_gain(weights, m, ky, kx0 + c, n_radii) * 0.0625f;
        const float2 v = t0[c][ky];
        t1[c][__brev((unsigned)ky) >> 24] = make_float2(v.x * g, v.y * g);
    }
    fft8lines(t1, t, +1.0f);                 // inverse along columns
    #pragma unroll
    for (int i = 0; i < 8; ++i) {
        const int idx = t + i * 256;
        const int ky = idx >> 3, c = idx & 7;
        const float2 v = t1[c][ky];
        spec[base + ((size_t)ky << 8) + kx0 + c] =
            make_float2(v.x * 0.0625f, v.y * 0.0625f);
    }
}

// ---------------------------------------------------------------------------
// Kernel 4: inverse FFT along rows, keep real part -> x_enh
// ---------------------------------------------------------------------------
__global__ void __launch_bounds__(256)
fft_rows_inv_kernel(const float2* __restrict__ spec, float* __restrict__ xenh) {
    __shared__ float2 tile[8][256];
    const int blk   = blockIdx.x;
    const int plane = blk >> 5;
    const int row0  = (blk & 31) * 8;
    const size_t base = ((size_t)plane << 16) + ((size_t)row0 << 8);
    const int t = threadIdx.x;

    #pragma unroll
    for (int i = 0; i < 8; ++i) {
        const int idx = t + i * 256;
        const int l = idx >> 8, k = idx & 255;
        tile[l][__brev((unsigned)k) >> 24] = spec[base + idx];
    }
    fft8lines(tile, t, +1.0f);
    #pragma unroll
    for (int i = 0; i < 8; ++i) {
        const int idx = t + i * 256;
        const int l = idx >> 8, k = idx & 255;
        xenh[base + idx] = tile[l][k].x * 0.0625f;
    }
}

// ---------------------------------------------------------------------------
// Kernel 5: out[b,c,hw] = x[b,c,hw] + sum_m w_out[c,m] * xenh[b,m,hw]
// One wave per 16-pixel tile: load B (xenh) fragment ONCE, loop 16 c-tiles
// reusing it -> 16 WMMAs per wave, xenh read exactly once from memory.
// ---------------------------------------------------------------------------
__global__ void __launch_bounds__(256)
proj_out_kernel(const float* __restrict__ xenh, const float* __restrict__ w_out,
                const float* __restrict__ x, float* __restrict__ out) {
    const int tid    = threadIdx.x;
    const int lane   = tid & 31;
    const int wave   = tid >> 5;
    const int ptile  = blockIdx.x * 8 + wave;      // 16384 pixel tiles
    const int row    = lane & 15;
    const int laneHi = lane >> 4;
    const int p      = ptile * 16 + row;
    const int b      = p >> 16;
    const int hw     = p & 65535;

    v16bf Bm = {};
    #pragma unroll
    for (int v = 0; v < 8; ++v) {
        #pragma unroll
        for (int h = 0; h < 2; ++h) {
            const int k = kmap(v, h, laneHi);
            if (k < 16)                            // K=16 valid, rest zero-pad
                Bm[2 * v + h] = (__bf16)xenh[(((size_t)b * 16 + k) << 16) + hw];
        }
    }

    const int clsub = laneHi << 3;
    #pragma unroll
    for (int ct = 0; ct < 16; ++ct) {
        v16bf A = {};
        #pragma unroll
        for (int v = 0; v < 8; ++v) {
            #pragma unroll
            for (int h = 0; h < 2; ++h) {
                const int k = kmap(v, h, laneHi);
                if (k < 16)
                    A[2 * v + h] = (__bf16)w_out[(ct * 16 + row) * 16 + k];
            }
        }
        v8f acc = {};
        acc = __builtin_amdgcn_wmma_f32_16x16x32_bf16(
            false, A, false, Bm, (short)0, acc, false, false);

        #pragma unroll
        for (int d = 0; d < 8; ++d) {
            const size_t o = (((size_t)b * 256 + ct * 16 + clsub + d) << 16) + hw;
            out[o] = x[o] + acc[d];
        }
    }
}

// ---------------------------------------------------------------------------
extern "C" void kernel_launch(void* const* d_in, const int* in_sizes, int n_in,
                              void* d_out, int out_size, void* d_ws, size_t ws_size,
                              hipStream_t stream) {
    (void)n_in; (void)out_size; (void)ws_size;
    const float* x       = (const float*)d_in[0];   // [4,256,256,256]
    const float* w_in    = (const float*)d_in[1];   // [16,256]
    const float* w_out   = (const float*)d_in[2];   // [256,16]
    const float* weights = (const float*)d_in[3];   // [16,8,n_radii]
    float* out = (float*)d_out;

    const int n_radii = in_sizes[3] / (16 * 8);     // 17

    // workspace carve: xp 16 MiB | spec 32 MiB | xenh 16 MiB
    float*  xp   = (float*)d_ws;
    float2* spec = (float2*)((char*)d_ws + (size_t)16 * 1024 * 1024);
    float*  xenh = (float*)((char*)d_ws + (size_t)48 * 1024 * 1024);

    dim3 blk(256);
    proj_in_kernel      <<<2048, blk, 0, stream>>>(x, w_in, xp);
    fft_rows_fwd_kernel <<<2048, blk, 0, stream>>>(xp, spec);
    fft_cols_gain_kernel<<<2048, blk, 0, stream>>>(spec, weights, n_radii);
    fft_rows_inv_kernel <<<2048, blk, 0, stream>>>(spec, xenh);
    proj_out_kernel     <<<2048, blk, 0, stream>>>(xenh, w_out, x, out);
}
